// TwoLayerMLP_62448824484605
// MI455X (gfx1250) — matrix operand order repaired and checked by reading, not compiled, not run on hardware
//
#include <hip/hip_runtime.h>

// Attention: out = softmax(Q K^T / sqrt(D)) V
// B=2 H=8 Q=2048 N=4096 D=128, fp32 in/out, bf16 WMMA compute (f32 accum).
// Flash-attention, wave32, double-buffered LDS tiles, 17 WMMAs / 32-key step.

#define BH_  16
#define Q_   2048
#define N_   4096
#define D_   128

typedef __attribute__((ext_vector_type(16))) __bf16          v16bf;
typedef __attribute__((ext_vector_type(2)))  __bf16          v2bf;
typedef __attribute__((ext_vector_type(8)))  float           v8f;
typedef __attribute__((ext_vector_type(8)))  unsigned short  u16x8;
typedef __attribute__((ext_vector_type(16))) unsigned short  u16x16;

union BFVec { u16x16 u; v16bf b; };

static __device__ __forceinline__ unsigned short f2bf(float f) {
    unsigned u = __float_as_uint(f);
    u += 0x7FFFu + ((u >> 16) & 1u);      // round-to-nearest-even
    return (unsigned short)(u >> 16);
}

static __device__ __forceinline__ unsigned pack2bf(float a, float b) {
#if __has_builtin(__builtin_amdgcn_cvt_pk_bf16_f32)
    v2bf t = __builtin_amdgcn_cvt_pk_bf16_f32(a, b);
    return __builtin_bit_cast(unsigned, t);
#else
    return (unsigned)f2bf(a) | ((unsigned)f2bf(b) << 16);
#endif
}

// max-reduce across each 16-lane half (DPP row rotates; VALU, not DS pipe)
static __device__ __forceinline__ float rmax16(float x) {
#if __has_builtin(__builtin_amdgcn_update_dpp)
    x = fmaxf(x, __int_as_float(__builtin_amdgcn_update_dpp(
                 0, __float_as_int(x), 0x128, 0xF, 0xF, true)));   // row_ror:8
    x = fmaxf(x, __int_as_float(__builtin_amdgcn_update_dpp(
                 0, __float_as_int(x), 0x124, 0xF, 0xF, true)));   // row_ror:4
    x = fmaxf(x, __int_as_float(__builtin_amdgcn_update_dpp(
                 0, __float_as_int(x), 0x122, 0xF, 0xF, true)));   // row_ror:2
    x = fmaxf(x, __int_as_float(__builtin_amdgcn_update_dpp(
                 0, __float_as_int(x), 0x121, 0xF, 0xF, true)));   // row_ror:1
#else
    x = fmaxf(x, __shfl_xor(x, 1, 32));
    x = fmaxf(x, __shfl_xor(x, 2, 32));
    x = fmaxf(x, __shfl_xor(x, 4, 32));
    x = fmaxf(x, __shfl_xor(x, 8, 32));
#endif
    return x;
}

static __device__ __forceinline__ v16bf cat16(u16x8 lo, u16x8 hi) {
    BFVec t;
    t.u = __builtin_shufflevector(lo, hi, 0,1,2,3,4,5,6,7,8,9,10,11,12,13,14,15);
    return t.b;
}

// LDS row pitches (ushorts), padded for conflict-free b128 fragment reads
#define KROW 136   // 128 + 8  (row = 272B, 16B aligned)
#define VROW 40    // 32 + 8   (row = 80B,  16B aligned)
#define PROW 40    // 32 + 8

// stage one 32-key K/V tile: global fp32 -> LDS bf16 (K row-major, V transposed)
static __device__ __forceinline__ void stage_tile(
    unsigned short* __restrict__ ksb, unsigned short* __restrict__ vtb,
    const float* __restrict__ kSrc, const float* __restrict__ vSrc, int tid)
{
#pragma unroll
    for (int i = 0; i < 4; ++i) {
        int idx = tid + i * 256;          // float4 index into 32x128 tile
        int row = idx >> 5;
        int c4  = idx & 31;
        const float4 kv = ((const float4*)(kSrc + (size_t)row * D_))[c4];
        unsigned long long pk =
              (unsigned long long)pack2bf(kv.x, kv.y)
            | ((unsigned long long)pack2bf(kv.z, kv.w) << 32);
        *(unsigned long long*)(ksb + row * KROW + c4 * 4) = pk;

        const float4 vv = ((const float4*)(vSrc + (size_t)row * D_))[c4];
        vtb[(c4 * 4 + 0) * VROW + row] = f2bf(vv.x);
        vtb[(c4 * 4 + 1) * VROW + row] = f2bf(vv.y);
        vtb[(c4 * 4 + 2) * VROW + row] = f2bf(vv.z);
        vtb[(c4 * 4 + 3) * VROW + row] = f2bf(vv.w);
    }
}

__launch_bounds__(256)
__global__ void fa_wmma_bf16_kernel(const float* __restrict__ Qm,
                                    const float* __restrict__ Km,
                                    const float* __restrict__ Vm,
                                    float* __restrict__ Om)
{
    __shared__ unsigned short Ks [2][32  * KROW];  // K tiles, bf16 [key][d]
    __shared__ unsigned short Vts[2][128 * VROW];  // V tiles, bf16 transposed [d][key]
    __shared__ unsigned short Ps [8 * 16 * PROW];  // per-wave P staging [16][32]

    const int bh   = blockIdx.y;
    const int q0   = blockIdx.x * 128;
    const int tid  = threadIdx.x;
    const int wid  = tid >> 5;
    const int lane = tid & 31;
    const int h    = lane >> 4;   // half-wave
    const int ln   = lane & 15;

    const float scale = 0.08838834764831845f;    // 1/sqrt(128)

    // ---- preload Q A-fragments (scaled, bf16): 4 chunks of 32 d each ----
    const int qrow = q0 + wid * 16 + ln;
    const float* qp = Qm + ((size_t)bh * Q_ + qrow) * D_;
    v16bf qa[4];
#pragma unroll
    for (int c = 0; c < 4; ++c) {
        const float* p0 = qp + c * 32 + 16 * h;       // k = 16h + i
        const float* p1 = qp + c * 32 + 16 * h + 8;   // k = 16h + 8 + i
        BFVec t;
#pragma unroll
        for (int i = 0; i < 8; ++i) t.u[i]     = f2bf(p0[i] * scale);
#pragma unroll
        for (int i = 0; i < 8; ++i) t.u[8 + i] = f2bf(p1[i] * scale);
        qa[c] = t.b;
    }

    // constant all-ones B fragment: row-sum of P via the WMMA pipe
    v16bf onesb;
    {
        BFVec t;
#pragma unroll
        for (int i = 0; i < 16; ++i) t.u[i] = 0x3F80;  // bf16 1.0
        onesb = t.b;
    }

    // ---- state ----
    v8f acc[8];
#pragma unroll
    for (int c = 0; c < 8; ++c)
#pragma unroll
        for (int r = 0; r < 8; ++r) acc[c][r] = 0.0f;
    v8f lacc;
    float mOld[8];
#pragma unroll
    for (int r = 0; r < 8; ++r) { mOld[r] = -3.0e38f; lacc[r] = 0.0f; }

    const float* kBase = Km + (size_t)bh * N_ * D_;
    const float* vBase = Vm + (size_t)bh * N_ * D_;
    unsigned short* pw = Ps + wid * 16 * PROW;

    // prologue: stage tile 0 into buffer 0
    stage_tile(Ks[0], Vts[0], kBase, vBase, tid);
    __syncthreads();

    const int NIT = N_ / 32;
    for (int it = 0; it < NIT; ++it) {
        const int cur = it & 1;

        // ---- stage NEXT tile into the other buffer (overlaps with compute) ----
        if (it + 1 < NIT)
            stage_tile(Ks[cur ^ 1], Vts[cur ^ 1],
                       kBase + (size_t)(it + 1) * 32 * D_,
                       vBase + (size_t)(it + 1) * 32 * D_, tid);
        if (it + 2 < NIT) {   // prefetch tile it+2 (global_prefetch_b8)
            __builtin_prefetch(kBase + (size_t)(it + 2) * 32 * D_ + tid * 16, 0, 1);
            __builtin_prefetch(vBase + (size_t)(it + 2) * 32 * D_ + tid * 16, 0, 1);
        }

        const unsigned short* ksb = Ks[cur];
        const unsigned short* vtb = Vts[cur];

        // ---- S = Q K^T  (two 16x16 tiles: keys 0-15 and 16-31) ----
        v8f s0, s1;
#pragma unroll
        for (int r = 0; r < 8; ++r) { s0[r] = 0.0f; s1[r] = 0.0f; }
#pragma unroll
        for (int c = 0; c < 4; ++c) {
            const unsigned short* kr0 = ksb + ln * KROW        + c * 32 + 16 * h;
            const unsigned short* kr1 = ksb + (16 + ln) * KROW + c * 32 + 16 * h;
            v16bf b0 = cat16(*(const u16x8*)kr0, *(const u16x8*)(kr0 + 8));
            v16bf b1 = cat16(*(const u16x8*)kr1, *(const u16x8*)(kr1 + 8));
            s0 = __builtin_amdgcn_wmma_f32_16x16x32_bf16(false, qa[c], false, b0,
                                                         (short)0, s0, false, false);
            s1 = __builtin_amdgcn_wmma_f32_16x16x32_bf16(false, qa[c], false, b1,
                                                         (short)0, s1, false, false);
        }

        // ---- online softmax (row m = r + 8h; stats replicated per 16-lane half) ----
        float alpha[8];
#pragma unroll
        for (int r = 0; r < 8; ++r) {
            float t  = rmax16(fmaxf(s0[r], s1[r]));
            float mN = fmaxf(mOld[r], t);
            alpha[r] = __expf(mOld[r] - mN);
            mOld[r]  = mN;
            s0[r] = __expf(s0[r] - mN);
            s1[r] = __expf(s1[r] - mN);
        }
#pragma unroll
        for (int c = 0; c < 8; ++c)
#pragma unroll
            for (int r = 0; r < 8; ++r) acc[c][r] *= alpha[r];
#pragma unroll
        for (int r = 0; r < 8; ++r) lacc[r] *= alpha[r];

        // ---- P -> bf16, restage via per-wave LDS into A-fragment layout ----
#pragma unroll
        for (int r = 0; r < 8; ++r) {
            pw[(r + 8 * h) * PROW + ln]      = f2bf(s0[r]);
            pw[(r + 8 * h) * PROW + 16 + ln] = f2bf(s1[r]);
        }
        const unsigned short* pr = pw + ln * PROW + 16 * h;
        v16bf pa = cat16(*(const u16x8*)pr, *(const u16x8*)(pr + 8));

        // row-sum of P on the WMMA pipe: l accumulates like a 129th output column
        lacc = __builtin_amdgcn_wmma_f32_16x16x32_bf16(false, pa, false, onesb,
                                                       (short)0, lacc, false, false);

        // ---- O += P V  (8 d-chunks of 16) ----
#pragma unroll
        for (int c = 0; c < 8; ++c) {
            const unsigned short* vr = vtb + (c * 16 + ln) * VROW + 16 * h;
            v16bf vb = cat16(*(const u16x8*)vr, *(const u16x8*)(vr + 8));
            acc[c] = __builtin_amdgcn_wmma_f32_16x16x32_bf16(false, pa, false, vb,
                                                             (short)0, acc[c], false, false);
        }

        __syncthreads();   // next-tile staging visible; current buffer free to reuse
    }

    // ---- normalize and store ----
    float inv[8];
#pragma unroll
    for (int r = 0; r < 8; ++r) inv[r] = 1.0f / lacc[r];
    float* ob = Om + ((size_t)bh * Q_ + q0 + wid * 16 + 8 * h) * D_ + ln;
#pragma unroll
    for (int c = 0; c < 8; ++c)
#pragma unroll
        for (int r = 0; r < 8; ++r)
            ob[(size_t)r * D_ + c * 16] = acc[c][r] * inv[r];
}

extern "C" void kernel_launch(void* const* d_in, const int* in_sizes, int n_in,
                              void* d_out, int out_size, void* d_ws, size_t ws_size,
                              hipStream_t stream) {
    (void)in_sizes; (void)n_in; (void)out_size; (void)d_ws; (void)ws_size;
    const float* q = (const float*)d_in[0];
    const float* k = (const float*)d_in[1];
    const float* v = (const float*)d_in[2];
    float* o = (float*)d_out;
    dim3 grid(Q_ / 128, BH_);
    dim3 block(256);
    fa_wmma_bf16_kernel<<<grid, block, 0, stream>>>(q, k, v, o);
}
